// ElasticAttentionBlock_558345748857
// MI455X (gfx1250) — matrix operand order repaired and checked by reading, not compiled, not run on hardware
//
#include <hip/hip_runtime.h>
#include <hip/hip_bf16.h>

typedef __attribute__((ext_vector_type(16))) __bf16 v16bf;
typedef __attribute__((ext_vector_type(8)))  __bf16 v8bf;
typedef __attribute__((ext_vector_type(8)))  float  v8f;
typedef __attribute__((ext_vector_type(4)))  int    i32x4;

#define NQ  4096
#define NK  8192
#define DIM 1024

// -------------------- gfx1250 async global->LDS support --------------------
#if defined(__has_builtin)
#if __has_builtin(__builtin_amdgcn_global_load_async_to_lds_b128)
#define HAVE_ASYNC 1
#endif
#endif
#ifndef HAVE_ASYNC
#define HAVE_ASYNC 0
#endif

#if HAVE_ASYNC
// builtin signature: (i32x4 addrspace(1)*, i32x4 addrspace(3)*, imm offset, imm cpol)
typedef __attribute__((address_space(1))) i32x4* gp128;
typedef __attribute__((address_space(3))) i32x4* lp128;
__device__ __forceinline__ void async_cp16(const void* g, void* l) {
    // generic LDS pointer's low 32 bits are the LDS offset (aperture truncation)
    __builtin_amdgcn_global_load_async_to_lds_b128(
        (gp128)(size_t)g, (lp128)(unsigned int)(size_t)l, 0, 0);
}
#endif

template <int N>
__device__ __forceinline__ void wait_async() {
#if defined(__has_builtin) && __has_builtin(__builtin_amdgcn_s_wait_asynccnt)
    __builtin_amdgcn_s_wait_asynccnt(N);
#else
    asm volatile("s_wait_asynccnt %0" ::"i"(N) : "memory");
#endif
}

// ---------------------------------------------------------------------------
// Kernel 1: per-row L2 normalization of Q and K, f32 -> bf16
// ---------------------------------------------------------------------------
struct alignas(8) bh4 { __hip_bfloat16 a, b, c, d; };

__global__ __launch_bounds__(256)
void prep_norms(const float* __restrict__ Q, const float* __restrict__ K,
                __hip_bfloat16* __restrict__ Qn, __hip_bfloat16* __restrict__ Kn)
{
    const int row = blockIdx.x;
    const bool isQ = (row < NQ);
    const float* src = isQ ? (Q + (size_t)row * DIM)
                           : (K + (size_t)(row - NQ) * DIM);
    const int t = threadIdx.x;

    float4 x = *(const float4*)(src + t * 4);
    float ss = x.x * x.x + x.y * x.y + x.z * x.z + x.w * x.w;
#pragma unroll
    for (int off = 16; off > 0; off >>= 1)
        ss += __shfl_xor(ss, off, 32);

    __shared__ float wsum[8];
    if ((t & 31) == 0) wsum[t >> 5] = ss;
    __syncthreads();
    float tot = 0.f;
#pragma unroll
    for (int i = 0; i < 8; i++) tot += wsum[i];
    const float inv = rsqrtf(tot);

    __hip_bfloat16* dst = isQ ? (Qn + (size_t)row * DIM)
                              : (Kn + (size_t)(row - NQ) * DIM);
    bh4 o;
    o.a = __float2bfloat16(x.x * inv);
    o.b = __float2bfloat16(x.y * inv);
    o.c = __float2bfloat16(x.z * inv);
    o.d = __float2bfloat16(x.w * inv);
    *(bh4*)(dst + t * 4) = o;
}

// ---------------------------------------------------------------------------
// Kernel 2: transpose + convert V [8192,1024] f32 -> VbT [1024,8192] bf16
// ---------------------------------------------------------------------------
__global__ __launch_bounds__(256)
void transpose_conv(const float* __restrict__ V, __hip_bfloat16* __restrict__ VbT)
{
    __shared__ float tile[64][65];
    const int k0 = blockIdx.x * 64;
    const int d0 = blockIdx.y * 64;
    const int t  = threadIdx.x;
#pragma unroll
    for (int i = 0; i < 16; i++) {
        int idx = t + i * 256;
        int r = idx >> 6, c = idx & 63;          // r: k, c: d
        tile[r][c] = V[(size_t)(k0 + r) * DIM + d0 + c];
    }
    __syncthreads();
#pragma unroll
    for (int i = 0; i < 16; i++) {
        int idx = t + i * 256;
        int r = idx >> 6, c = idx & 63;          // r: d, c: k
        VbT[(size_t)(d0 + r) * NK + k0 + c] = __float2bfloat16(tile[c][r]);
    }
}

// ---------------------------------------------------------------------------
// Kernel 3: NT GEMM bf16->f32:  C[M,N] = Arows[M,Kd] * Brows[N,Kd]^T
// 8 waves, block tile TBM x TBN x 32, wave grid 2(M) x 4(N),
// wave tile (MI*16) x (NJ*16). Double-buffered LDS; async staging if present.
// ---------------------------------------------------------------------------
#define BK 32
#define LP 40   // padded LDS row (bf16): 80B, keeps 16B alignment, spreads banks

template <int TBM, int TBN, int MI, int NJ>
__global__ __launch_bounds__(256)
void gemm_nt_bf16(const __hip_bfloat16* __restrict__ A, int lda,
                  const __hip_bfloat16* __restrict__ B, int ldb,
                  float* __restrict__ C, int ldc, int Kd)
{
    constexpr int CHA = TBM * BK / (8 * 256);   // b128 chunks/thread for A tile
    constexpr int CHB = TBN * BK / (8 * 256);
    constexpr int NLOADS = CHA + CHB;

    __shared__ __hip_bfloat16 As[2][TBM][LP];
    __shared__ __hip_bfloat16 Bs[2][TBN][LP];

    const int t    = threadIdx.x;
    const int wave = t >> 5;
    const int lane = t & 31;
    const int m0   = blockIdx.y * TBM;
    const int n0   = blockIdx.x * TBN;
    const int wm   = (wave & 1) * (MI * 16);
    const int wn   = (wave >> 1) * (NJ * 16);

    // A fragment: ISA 16-bit A layout (interleaved K chunks per half-wave)
    const int arow = lane & 15;
    const int ak0  = (lane >> 4) * 8;     // K {0..7,16..23} / {8..15,24..31}
    // B fragment: same interleaved K chunks as A
    const int bcol = lane & 15;
    const int bk0  = (lane >> 4) * 8;     // K {0..7,16..23} / {8..15,24..31}

    v8f acc[MI][NJ] = {};

#if HAVE_ASYNC
    auto stage_issue = [&](int buf, int kb) {
#pragma unroll
        for (int i = 0; i < CHA; ++i) {
            int c = t + i * 256, r = c >> 2, col = (c & 3) * 8;
            async_cp16(A + (size_t)(m0 + r) * lda + kb + col, &As[buf][r][col]);
        }
#pragma unroll
        for (int i = 0; i < CHB; ++i) {
            int c = t + i * 256, r = c >> 2, col = (c & 3) * 8;
            async_cp16(B + (size_t)(n0 + r) * ldb + kb + col, &Bs[buf][r][col]);
        }
    };
#else
    float4 ra[CHA], rb[CHB];
    auto stage_fetch = [&](int kb) {
#pragma unroll
        for (int i = 0; i < CHA; ++i) {
            int c = t + i * 256, r = c >> 2, col = (c & 3) * 8;
            ra[i] = *(const float4*)(A + (size_t)(m0 + r) * lda + kb + col);
        }
#pragma unroll
        for (int i = 0; i < CHB; ++i) {
            int c = t + i * 256, r = c >> 2, col = (c & 3) * 8;
            rb[i] = *(const float4*)(B + (size_t)(n0 + r) * ldb + kb + col);
        }
    };
    auto stage_store = [&](int buf) {
#pragma unroll
        for (int i = 0; i < CHA; ++i) {
            int c = t + i * 256, r = c >> 2, col = (c & 3) * 8;
            *(float4*)&As[buf][r][col] = ra[i];
        }
#pragma unroll
        for (int i = 0; i < CHB; ++i) {
            int c = t + i * 256, r = c >> 2, col = (c & 3) * 8;
            *(float4*)&Bs[buf][r][col] = rb[i];
        }
    };
#endif

    const int steps = Kd / BK;
#if HAVE_ASYNC
    stage_issue(0, 0);
#else
    stage_fetch(0);
    stage_store(0);
    __syncthreads();
#endif

    for (int s = 0; s < steps; ++s) {
        const int buf = s & 1;
        const int kbn = (s + 1) * BK;
#if HAVE_ASYNC
        if (s + 1 < steps) { stage_issue(buf ^ 1, kbn); wait_async<NLOADS>(); }
        else               { wait_async<0>(); }
        __syncthreads();
#else
        if (s + 1 < steps) stage_fetch(kbn);     // overlap with compute below
#endif

        v16bf af[MI], bfr[NJ];
#pragma unroll
        for (int i = 0; i < MI; i++) {
            const __hip_bfloat16* p = &As[buf][wm + i * 16 + arow][ak0];
            v8bf lo = *(const v8bf*)(p);
            v8bf hi = *(const v8bf*)(p + 16);
            af[i] = __builtin_shufflevector(lo, hi, 0,1,2,3,4,5,6,7,8,9,10,11,12,13,14,15);
        }
#pragma unroll
        for (int j = 0; j < NJ; j++) {
            const __hip_bfloat16* p = &Bs[buf][wn + j * 16 + bcol][bk0];
            v8bf lo = *(const v8bf*)(p);
            v8bf hi = *(const v8bf*)(p + 16);
            bfr[j] = __builtin_shufflevector(lo, hi, 0,1,2,3,4,5,6,7,8,9,10,11,12,13,14,15);
        }
#pragma unroll
        for (int i = 0; i < MI; i++)
#pragma unroll
            for (int j = 0; j < NJ; j++)
                acc[i][j] = __builtin_amdgcn_wmma_f32_16x16x32_bf16(
                    false, af[i], false, bfr[j], (short)0, acc[i][j], false, false);

#if HAVE_ASYNC
        __syncthreads();
#else
        if (s + 1 < steps) stage_store(buf ^ 1);
        __syncthreads();
#endif
    }

    // epilogue: C tile layout — VGPR r: M = r + 8*(lane/16), N = lane%16
#pragma unroll
    for (int i = 0; i < MI; i++)
#pragma unroll
        for (int j = 0; j < NJ; j++) {
            const int mrow = m0 + wm + i * 16 + (lane >> 4) * 8;
            const int ncol = n0 + wn + j * 16 + (lane & 15);
            float* cp = C + (size_t)mrow * ldc + ncol;
#pragma unroll
            for (int r = 0; r < 8; r++) cp[(size_t)r * ldc] = acc[i][j][r];
        }
}

// ---------------------------------------------------------------------------
// Kernel 4: row softmax on S [4096,8192] f32, write bf16 P IN PLACE
// (P keeps the f32 row pitch: element pitch 2*NK). Reads precede the
// reduction barriers; writes follow them -> alias-safe.
// ---------------------------------------------------------------------------
struct alignas(8) bh4s { __hip_bfloat16 a, b, c, d; };

__global__ __launch_bounds__(256)
void softmax_rows(float* __restrict__ S)
{
    const int row = blockIdx.x;
    float* Srow = S + (size_t)row * NK;
    const int t = threadIdx.x;

    float4 v[8];
    float mx = -3.4e38f;
#pragma unroll
    for (int i = 0; i < 8; i++) {
        v[i] = *(const float4*)(Srow + t * 4 + i * 1024);
        mx = fmaxf(mx, fmaxf(fmaxf(v[i].x, v[i].y), fmaxf(v[i].z, v[i].w)));
    }
#pragma unroll
    for (int off = 16; off > 0; off >>= 1)
        mx = fmaxf(mx, __shfl_xor(mx, off, 32));
    __shared__ float red[8];
    if ((t & 31) == 0) red[t >> 5] = mx;
    __syncthreads();
#pragma unroll
    for (int i = 0; i < 8; i++) mx = fmaxf(mx, red[i]);
    __syncthreads();

    const float l2e = 1.4426950408889634f;
    float sum = 0.f;
#pragma unroll
    for (int i = 0; i < 8; i++) {
        v[i].x = exp2f((v[i].x - mx) * l2e);
        v[i].y = exp2f((v[i].y - mx) * l2e);
        v[i].z = exp2f((v[i].z - mx) * l2e);
        v[i].w = exp2f((v[i].w - mx) * l2e);
        sum += v[i].x + v[i].y + v[i].z + v[i].w;
    }
#pragma unroll
    for (int off = 16; off > 0; off >>= 1)
        sum += __shfl_xor(sum, off, 32);
    if ((t & 31) == 0) red[t >> 5] = sum;
    __syncthreads();
    float tot = 0.f;
#pragma unroll
    for (int i = 0; i < 8; i++) tot += red[i];
    const float inv = 1.0f / tot;

    __hip_bfloat16* Prow = (__hip_bfloat16*)S + (size_t)row * (2 * NK);
#pragma unroll
    for (int i = 0; i < 8; i++) {
        bh4s o;
        o.a = __float2bfloat16(v[i].x * inv);
        o.b = __float2bfloat16(v[i].y * inv);
        o.c = __float2bfloat16(v[i].z * inv);
        o.d = __float2bfloat16(v[i].w * inv);
        *(bh4s*)(Prow + t * 4 + i * 1024) = o;
    }
}

// ---------------------------------------------------------------------------
// Launch: prep -> transpose -> GEMM1 (scores) -> softmax -> GEMM2 (context)
// Workspace: S f32 128MB | Qn 8MB | Kn 16MB | VbT 16MB  (168MB total)
// ---------------------------------------------------------------------------
extern "C" void kernel_launch(void* const* d_in, const int* in_sizes, int n_in,
                              void* d_out, int out_size, void* d_ws, size_t ws_size,
                              hipStream_t stream)
{
    const float* Q = (const float*)d_in[0];   // [4096,1024]
    const float* K = (const float*)d_in[1];   // [8192,1024]
    float* out = (float*)d_out;               // [4096,1024]

    char* w = (char*)d_ws;
    float* S = (float*)w;                                               // 128 MB
    __hip_bfloat16* Qn  = (__hip_bfloat16*)(w + (size_t)NQ * NK * 4);   // 8 MB
    __hip_bfloat16* Kn  = Qn + (size_t)NQ * DIM;                        // 16 MB
    __hip_bfloat16* VbT = Kn + (size_t)NK * DIM;                        // 16 MB

    prep_norms<<<NQ + NK, 256, 0, stream>>>(Q, K, Qn, Kn);
    transpose_conv<<<dim3(NK / 64, DIM / 64), 256, 0, stream>>>(K, VbT);

    // S = Qn (4096x1024) * Kn^T -> block tile 128x256, wave tile 64x64
    gemm_nt_bf16<128, 256, 4, 4><<<dim3(NK / 256, NQ / 128), 256, 0, stream>>>(
        Qn, DIM, Kn, DIM, S, NK, DIM);

    softmax_rows<<<NQ, 256, 0, stream>>>(S);

    // out = P (4096x8192 bf16 in-place, pitch 2*NK) * VbT^T -> 128x128 tiles
    gemm_nt_bf16<128, 128, 4, 2><<<dim3(DIM / 128, NQ / 128), 256, 0, stream>>>(
        (const __hip_bfloat16*)w, 2 * NK, VbT, NK, out, DIM, NK);
}
